// PartialFC_V2_48447231099189
// MI455X (gfx1250) — matrix operand order repaired and checked by reading, not compiled, not run on hardware
//
#include <hip/hip_runtime.h>
#include <hip/hip_bf16.h>
#include <math.h>
#include <stdint.h>

// ---------------------------------------------------------------------------
// ArcFace / PartialFC for MI455X (gfx1250, wave32, WMMA).
// B=256, D=512, C=100000, RD=8.
// ---------------------------------------------------------------------------

#define B_ROWS 256
#define D_DIM  512
#define C_CLS  100000
#define RD     8

#define SCALE_S 64.0f
// cos(0.5), sin(0.5), cos(pi-0.5) = -cos(0.5), sin(pi-0.5)*0.5
#define COS_M   0.8775825618903728f
#define SIN_M   0.479425538604203f
#define THRESH  (-0.8775825618903728f)
#define MM_FALL 0.2397127693021015f

typedef __attribute__((ext_vector_type(16))) __bf16 v16bf;
typedef __attribute__((ext_vector_type(8)))  float  v8f;

union FragA { v16bf bf; uint4 q[2]; };
union FragB { v16bf bf; __hip_bfloat162 h[8]; };

__device__ __forceinline__ uint32_t bf16_bits(float f) {
    uint32_t u = __float_as_uint(f);
    return (u + 0x7FFFu + ((u >> 16) & 1u)) >> 16;   // round-to-nearest-even
}

// ---------------------------------------------------------------------------
// Kernel 1: L2-normalize embedding rows, store bf16 (row-major [256,512]).
// ---------------------------------------------------------------------------
__global__ void k_norm_embed(const float* __restrict__ emb,
                             unsigned short* __restrict__ abf) {
    const int row = blockIdx.x;          // 256 blocks
    const int t   = threadIdx.x;         // 256 threads
    const float* e = emb + (size_t)row * D_DIM;
    float x0 = e[t], x1 = e[t + 256];
    __shared__ float red[256];
    red[t] = x0 * x0 + x1 * x1;
    __syncthreads();
    for (int s = 128; s > 0; s >>= 1) {
        if (t < s) red[t] += red[t + s];
        __syncthreads();
    }
    float rinv = rsqrtf(red[0] + 1e-12f);
    abf[(size_t)row * D_DIM + t]       = (unsigned short)bf16_bits(x0 * rinv);
    abf[(size_t)row * D_DIM + t + 256] = (unsigned short)bf16_bits(x1 * rinv);
}

// ---------------------------------------------------------------------------
// Kernel 2: threefry2x32 PRNG (JAX default, key(42) -> (0,42)), stable
// argsort of 8 uniforms per row, keep first 3 -> labels3[256][3].
// ---------------------------------------------------------------------------
__device__ __forceinline__ void threefry2x32(uint32_t k0, uint32_t k1,
                                             uint32_t x0, uint32_t x1,
                                             uint32_t& y0, uint32_t& y1) {
    const uint32_t R[8] = {13, 15, 26, 6, 17, 29, 16, 24};
    uint32_t ks[3] = {k0, k1, k0 ^ k1 ^ 0x1BD11BDAu};
    x0 += ks[0]; x1 += ks[1];
    #pragma unroll
    for (int round = 0; round < 20; ++round) {
        uint32_t r = R[(round & 3) + (((round >> 2) & 1) ? 4 : 0)];
        x0 += x1;
        x1 = (x1 << r) | (x1 >> (32 - r));
        x1 ^= x0;
        if ((round & 3) == 3) {
            uint32_t s = (uint32_t)(round >> 2);   // 0..4
            x0 += ks[(s + 1) % 3];
            x1 += ks[(s + 2) % 3] + (s + 1);
        }
    }
    y0 = x0; y1 = x1;
}

__global__ void k_labels(const int* __restrict__ labels_in,
                         int* __restrict__ labels3) {
    const int row = threadIdx.x;   // 1 block, 256 threads
    float noise[RD];
    #pragma unroll
    for (int c = 0; c < RD; ++c) {
        uint32_t idx = (uint32_t)(row * RD + c);   // flat index in (256,8)
        uint32_t j   = idx & 1023u;                // pair index
        uint32_t y0, y1;
        threefry2x32(0u, 42u, j, 1024u + j, y0, y1);
        uint32_t bits = (idx < 1024u) ? y0 : y1;
        noise[c] = __uint_as_float((bits >> 9) | 0x3F800000u) - 1.0f;
    }
    // stable selection of the 3 smallest (argsort ascending, index tiebreak)
    bool used[RD] = {false};
    #pragma unroll
    for (int k = 0; k < 3; ++k) {
        int   bi = -1;
        float bv = 2.0f;
        #pragma unroll
        for (int c = 0; c < RD; ++c) {
            if (!used[c] && noise[c] < bv) { bv = noise[c]; bi = c; }
        }
        used[bi] = true;
        labels3[row * 3 + k] = labels_in[row * RD + bi];
    }
}

// ---------------------------------------------------------------------------
// Kernel 3: WMMA GEMM  logits = clip(normA @ normW^T, -1, 1).
// One wave per 16-class tile; W streamed exactly once from HBM
// (f32 -> v_cvt_pk_bf16_f32 in VALU), column sum-of-squares fused into the
// K loop, rsqrt applied in epilogue. Full B=256 (16 M-tiles) kept in a
// 128-VGPR f32 accumulator block; launch_bounds(128,1) gives the wave the
// whole register file so nothing spills. W loads for k+1 and the A fragment
// for tile m+1 are double-buffered ahead of each WMMA for load/MMA overlap.
// ---------------------------------------------------------------------------
__global__ void __launch_bounds__(128, 1)
k_gemm_wmma(const unsigned short* __restrict__ abf,
            const float* __restrict__ weight,
            float* __restrict__ logits) {
    const int wave = blockIdx.x * 4 + (threadIdx.x >> 5);
    if (wave >= C_CLS / 16) return;               // wave-uniform: EXEC stays all-1
    const int lane = threadIdx.x & 31;
    const int n    = lane & 15;                   // class within tile
    const int half = lane >> 4;                   // half-wave id
    const int c0   = wave * 16;

    // lane's W stream: K = k0 + half*16 .. +16 of class (c0+n)
    const float* wptr = weight + (size_t)(c0 + n) * D_DIM + (half << 4);
    // lane's A base: row = m*16 + n, K = half*16 .. +16 contiguous
    const unsigned short* abase = abf + (size_t)n * D_DIM + half * 16;

    v8f acc[16];
    #pragma unroll
    for (int m = 0; m < 16; ++m) acc[m] = (v8f)(0.0f);
    float sumsq = 0.0f;

    // prologue: first W chunk in flight
    float4 wbuf[4];
    #pragma unroll
    for (int i = 0; i < 4; ++i) wbuf[i] = *(const float4*)(wptr + 4 * i);

    for (int k0 = 0; k0 < D_DIM; k0 += 32) {
        // pack current W chunk -> B fragment, accumulate sum of squares
        FragB fb;
        #pragma unroll
        for (int i = 0; i < 4; ++i) {
            float4 w4 = wbuf[i];
            sumsq += w4.x * w4.x + w4.y * w4.y + w4.z * w4.z + w4.w * w4.w;
            fb.h[2 * i]     = __float22bfloat162_rn(make_float2(w4.x, w4.y));
            fb.h[2 * i + 1] = __float22bfloat162_rn(make_float2(w4.z, w4.w));
        }
        // issue next k-step's W loads early (hidden under the 16 WMMAs)
        if (k0 + 32 < D_DIM) {
            const float* wnext = wptr + 32;
            __builtin_prefetch(wnext + 32, 0, 0);    // global_prefetch_b8
            #pragma unroll
            for (int i = 0; i < 4; ++i) wbuf[i] = *(const float4*)(wnext + 4 * i);
            wptr = wnext;
        }

        // A fragments double-buffered across the 16 M-tiles
        const unsigned short* ak = abase + k0;
        FragA fa;
        fa.q[0] = *(const uint4*)(ak);
        fa.q[1] = *(const uint4*)(ak + 8);
        #pragma unroll
        for (int m = 0; m < 16; ++m) {
            FragA fn;
            if (m < 15) {
                const unsigned short* an = ak + (size_t)(m + 1) * 16 * D_DIM;
                fn.q[0] = *(const uint4*)(an);
                fn.q[1] = *(const uint4*)(an + 8);
            }
            acc[m] = __builtin_amdgcn_wmma_f32_16x16x32_bf16(
                false, fa.bf, false, fb.bf, (short)0, acc[m], false, false);
            if (m < 15) fa = fn;
        }
    }

    // column norm: combine the two K-halves held by lane pairs (L, L^16)
    float tot  = sumsq + __shfl_xor(sumsq, 16, 32);
    float rinv = rsqrtf(tot + 1e-12f);

    #pragma unroll
    for (int m = 0; m < 16; ++m) {
        const int mbase = m * 16 + half * 8;      // D layout: M = v + 8*half
        #pragma unroll
        for (int v = 0; v < 8; ++v) {
            float c = fminf(1.0f, fmaxf(-1.0f, acc[m][v] * rinv));
            logits[(size_t)(mbase + v) * C_CLS + c0 + n] = c;
        }
    }
}

// ---------------------------------------------------------------------------
// Kernel 4: per-row online-softmax + ArcFace margin correction for 3 targets.
// ---------------------------------------------------------------------------
__global__ void k_zero_out(float* out) { out[0] = 0.0f; }

__global__ void k_loss(const float* __restrict__ logits,
                       const int* __restrict__ labels3,
                       float* __restrict__ out) {
    const int row = blockIdx.x;                   // 256 blocks
    const int t   = threadIdx.x;                  // 256 threads
    const float* lr = logits + (size_t)row * C_CLS;

    float m = -INFINITY, s = 0.0f;
    for (int i = t; i < C_CLS; i += 256) {
        float x  = SCALE_S * lr[i];
        float mn = fmaxf(m, x);
        s = s * __expf(m - mn) + __expf(x - mn);
        m = mn;
    }
    __shared__ float sM[256], sS[256];
    sM[t] = m; sS[t] = s;
    __syncthreads();
    for (int w = 128; w > 0; w >>= 1) {
        if (t < w) {
            float m2 = sM[t + w], s2 = sS[t + w];
            float M  = fmaxf(sM[t], m2);
            sS[t] = sS[t] * __expf(sM[t] - M) + s2 * __expf(m2 - M);
            sM[t] = M;
        }
        __syncthreads();
    }
    if (t == 0) {
        const float M = sM[0], S = sS[0];
        float lsum = 0.0f;
        #pragma unroll
        for (int k = 0; k < 3; ++k) {
            int   l  = labels3[row * 3 + k];
            float tv = lr[l];                               // already in [-1,1]
            float tc = fminf(1.0f - 1e-7f, fmaxf(-1.0f + 1e-7f, tv));
            float mg = tc * COS_M - sqrtf(fmaxf(0.0f, 1.0f - tc * tc)) * SIN_M;
            float ft = (tv > THRESH) ? mg : (tv - MM_FALL);
            float o  = SCALE_S * tv;
            float v  = SCALE_S * ft;
            // ft <= tv on the valid range, so M remains the row max.
            float Sk = S - __expf(o - M) + __expf(v - M);
            lsum += -(v - M - __logf(Sk));
        }
        atomicAdd(out, lsum * (1.0f / (3.0f * (float)B_ROWS)));
    }
}

// ---------------------------------------------------------------------------
// Launch
// ---------------------------------------------------------------------------
extern "C" void kernel_launch(void* const* d_in, const int* in_sizes, int n_in,
                              void* d_out, int out_size, void* d_ws, size_t ws_size,
                              hipStream_t stream) {
    const float* emb    = (const float*)d_in[0];   // [256,512]
    const float* weight = (const float*)d_in[1];   // [100000,512]
    const int*   labels = (const int*)d_in[2];     // [256,8]
    (void)in_sizes; (void)n_in; (void)out_size; (void)ws_size;

    // workspace layout
    unsigned short* abf     = (unsigned short*)d_ws;                    // 256 KB
    int*            labels3 = (int*)((char*)d_ws + 262144);             // 3 KB
    float*          logits  = (float*)((char*)d_ws + 266240);           // 102.4 MB

    float* out = (float*)d_out;

    k_norm_embed<<<B_ROWS, 256, 0, stream>>>(emb, abf);
    k_labels<<<1, B_ROWS, 0, stream>>>(labels, labels3);

    const int nwaves = C_CLS / 16;                       // 6250 class tiles
    const int blocks = (nwaves + 3) / 4;                 // 4 waves / block
    k_gemm_wmma<<<blocks, 128, 0, stream>>>(abf, weight, logits);

    k_zero_out<<<1, 1, 0, stream>>>(out);
    k_loss<<<B_ROWS, 256, 0, stream>>>(logits, labels3, out);
}
